// TreeConvolution_67808943669268
// MI455X (gfx1250) — compile-verified, matrix operand order checked by reading, not run on hardware
//
#include <hip/hip_runtime.h>

// ---------------------------------------------------------------------------
// Types for CDNA5 WMMA (gfx1250, wave32)
// ---------------------------------------------------------------------------
typedef __bf16 bhalf;
typedef __attribute__((ext_vector_type(8)))  __bf16 bf16x8;
typedef __attribute__((ext_vector_type(16))) __bf16 bf16x16;
typedef __attribute__((ext_vector_type(8)))  float  floatx8;

#define CAT16(lo, hi) __builtin_shufflevector((lo), (hi), 0,1,2,3,4,5,6,7,8,9,10,11,12,13,14,15)

// Problem constants
#define B_SZ     256
#define P_IN     128
#define NCOL     256      // N+1
#define NL       765      // 3*(NCOL-1)
#define ACT_STRIDE 512    // row stride of actT (max channel count)
#define NTHREADS 512      // 16 wave32 waves
#define NWAVES   16

// Dynamic LDS layout
#define ACT_BYTES   (ACT_STRIDE * NCOL * 2)   // 262144: actT[col][c], bf16
#define LIDX_BYTES  (768 * 4)                 // gather indexes (padded)
#define RED_BYTES   64
#define LBIAS_BYTES (512 * 4)                 // conv bias staged in LDS
#define SMEM_BYTES  (ACT_BYTES + LIDX_BYTES + RED_BYTES + LBIAS_BYTES)

// Workspace layout (bytes)
#define OFF_WB1  ((size_t)0)                       // 512*384  bf16 = 393216
#define OFF_WB2  ((size_t)393216)                  // 512*1536 bf16 = 1572864
#define OFF_WB3  ((size_t)1966080)                 // 256*1536 bf16 = 786432
#define OFF_POOL ((size_t)2752512)                 // 256*256  f32  = 262144
#define OFF_SC   ((size_t)3014656)                 // 256 * 256*512 bf16 = 64 MiB

// ---------------------------------------------------------------------------
// Weight pack: f32 [Cout][Cin][3] -> bf16 [Cout][k*Cin + c]   (K-major taps)
// ---------------------------------------------------------------------------
__global__ void pack_weights(const float* __restrict__ W, bhalf* __restrict__ Wb,
                             int Cout, int Cin) {
    int i = blockIdx.x * blockDim.x + threadIdx.x;
    int total = Cout * Cin * 3;
    if (i >= total) return;
    int o = i / (3 * Cin);
    int r = i - o * (3 * Cin);
    int k = r / Cin;
    int c = r - k * Cin;
    Wb[i] = (bhalf)W[(o * Cin + c) * 3 + k];
}

// ---------------------------------------------------------------------------
// Fused tree-conv x3 + standardize + leaky + maxpool. One WG per sample.
// actT[col][c] bf16 in LDS (transposed activation).
// Each wave computes a 64x64 macro-tile: 4x4 grid of 16x16 WMMA tiles,
// A fragments reused across 4 column tiles, B fragments across 4 row tiles.
// ---------------------------------------------------------------------------
__global__ __launch_bounds__(NTHREADS, 1)
void tree_conv_fused(const float* __restrict__ trees,
                     const int*   __restrict__ indexes,
                     const float* __restrict__ cb1,
                     const float* __restrict__ cb2,
                     const float* __restrict__ cb3,
                     const bhalf* __restrict__ Wb1,
                     const bhalf* __restrict__ Wb2,
                     const bhalf* __restrict__ Wb3,
                     bhalf* __restrict__ scratch,
                     float* __restrict__ pooled) {
    extern __shared__ __align__(16) unsigned char smem_raw[];
    bhalf* actT  = reinterpret_cast<bhalf*>(smem_raw);
    int*   lidx  = reinterpret_cast<int*>(smem_raw + ACT_BYTES);
    float* red   = reinterpret_cast<float*>(smem_raw + ACT_BYTES + LIDX_BYTES);
    float* lbias = reinterpret_cast<float*>(smem_raw + ACT_BYTES + LIDX_BYTES + RED_BYTES);

    const int b    = blockIdx.x;
    const int tid  = threadIdx.x;
    const int lane = tid & 31;
    const int wid  = tid >> 5;
    const int m    = lane & 15;   // row/col within 16x16 tile
    const int h    = lane >> 4;   // K-half selector

    bhalf* sc = scratch + (size_t)b * (NCOL * ACT_STRIDE);

    // gather indexes (pad to 768 so n==255 tile lanes read a harmless 0)
    for (int i = tid; i < 768; i += NTHREADS) {
        int v = 0;
        if (i < NL) {
            v = indexes[(size_t)b * NL + i];
            v = v < 0 ? 0 : (v > 255 ? 255 : v);
        }
        lidx[i] = v;
    }
    // trees[b] : [128][256] f32 -> actT[col][c] bf16
    for (int i = tid; i < P_IN * NCOL; i += NTHREADS) {
        int c = i >> 8, col = i & 255;
        actT[col * ACT_STRIDE + c] = (bhalf)trees[(size_t)b * P_IN * NCOL + i];
    }
    __syncthreads();

    for (int layer = 0; layer < 3; ++layer) {
        int Cin, Cout, sh;
        const bhalf* Wb;
        const float* bias;
        if (layer == 0)      { Cin = 128; Cout = 512; sh = 9; Wb = Wb1; bias = cb1; }
        else if (layer == 1) { Cin = 512; Cout = 512; sh = 9; Wb = Wb2; bias = cb2; }
        else                 { Cin = 512; Cout = 256; sh = 8; Wb = Wb3; bias = cb3; }
        const int K3 = 3 * Cin;
        // macro-tiles: 64 rows x 64 cols each; col macro count is always 4
        const int nmacro = (Cout >> 6) << 2;

        if (tid < 4) red[tid] = 0.0f;
        for (int i = tid; i < Cout; i += NTHREADS) lbias[i] = bias[i];
        __syncthreads();

        float lsum = 0.0f, lsq = 0.0f;

        for (int mb = wid; mb < nmacro; mb += NWAVES) {
            const int ro0 = (mb >> 2) << 6;   // macro row origin (output channel)
            const int no0 = (mb & 3) << 6;    // macro col origin (position)

            floatx8 acc[4][4];
            #pragma unroll
            for (int rt = 0; rt < 4; ++rt)
                #pragma unroll
                for (int ct = 0; ct < 4; ++ct)
                    acc[rt][ct] = floatx8{0.f,0.f,0.f,0.f,0.f,0.f,0.f,0.f};

            #pragma unroll
            for (int k = 0; k < 3; ++k) {
                // per-column-tile gathered rows of actT
                const bhalf* br[4];
                #pragma unroll
                for (int ct = 0; ct < 4; ++ct) {
                    const int n = no0 + ct * 16 + m;
                    br[ct] = actT + lidx[3 * n + k] * ACT_STRIDE + 8 * h;
                }
                const bhalf* ar = Wb + (size_t)(ro0 + m) * K3 + k * Cin + 8 * h;

                for (int cb = 0; cb < Cin; cb += 32) {
                    bf16x16 a[4];
                    #pragma unroll
                    for (int rt = 0; rt < 4; ++rt) {
                        const bhalf* p = ar + (size_t)rt * 16 * K3 + cb;
                        a[rt] = CAT16(*(const bf16x8*)(p), *(const bf16x8*)(p + 16));
                    }
                    #pragma unroll
                    for (int ct = 0; ct < 4; ++ct) {
                        const bhalf* q = br[ct] + cb;
                        bf16x16 bb = CAT16(*(const bf16x8*)(q), *(const bf16x8*)(q + 16));
                        #pragma unroll
                        for (int rt = 0; rt < 4; ++rt) {
                            acc[rt][ct] = __builtin_amdgcn_wmma_f32_16x16x32_bf16(
                                false, a[rt], false, bb, (short)0, acc[rt][ct], false, false);
                        }
                    }
                }
            }

            // store 16 tiles (skip padding column n==255), accumulate stats
            #pragma unroll
            for (int ct = 0; ct < 4; ++ct) {
                const int n = no0 + ct * 16 + m;
                if (n < 255) {
                    #pragma unroll
                    for (int rt = 0; rt < 4; ++rt) {
                        const int o0 = ro0 + rt * 16;
                        bf16x8 pack;
                        #pragma unroll
                        for (int v = 0; v < 8; ++v) {
                            float x = acc[rt][ct][v] + lbias[o0 + v + 8 * h];
                            lsum += x;
                            lsq  += x * x;
                            pack[v] = (bhalf)x;
                        }
                        *(bf16x8*)(sc + (size_t)n * Cout + o0 + 8 * h) = pack;
                    }
                }
            }
        }

        atomicAdd(&red[0], lsum);
        atomicAdd(&red[1], lsq);
        __threadfence_block();
        __syncthreads();

        if (tid == 0) {
            const float cnt = (float)(Cout * NCOL);   // includes the zero pad column
            float mu  = red[0] / cnt;
            float var = (red[1] - cnt * mu * mu) / (cnt - 1.0f);  // ddof=1 (torch.std)
            var = var > 0.0f ? var : 0.0f;
            red[2] = mu;
            red[3] = 1.0f / (sqrtf(var) + 1e-5f);
        }
        __syncthreads();

        const float mu  = red[2];
        const float inv = red[3];
        const int total = Cout * NCOL;
        for (int i = tid; i < total; i += NTHREADS) {
            int col = i >> sh;
            int c   = i & (Cout - 1);
            float x = (col == 0) ? 0.0f : (float)sc[(size_t)(col - 1) * Cout + c];
            float y = (x - mu) * inv;
            y = y > 0.0f ? y : 0.01f * y;                 // leaky_relu
            actT[col * ACT_STRIDE + c] = (bhalf)y;
        }
        __syncthreads();
    }

    // TreeMaxPool: max over 256 columns of the final [256 x 256] activation
    for (int c = tid; c < 256; c += NTHREADS) {
        float mx = -3.0e38f;
        for (int col = 0; col < NCOL; ++col) {
            float v = (float)actT[col * ACT_STRIDE + c];
            mx = mx > v ? mx : v;
        }
        pooled[(size_t)b * 256 + c] = mx;
    }
}

// ---------------------------------------------------------------------------
// Tiny out_mlp: per-sample Linear->LN->leaky x2 -> Linear.  f32 VALU.
// ---------------------------------------------------------------------------
__global__ __launch_bounds__(128)
void mlp_kernel(const float* __restrict__ pooled,
                const float* __restrict__ W1, const float* __restrict__ b1,
                const float* __restrict__ g1, const float* __restrict__ be1,
                const float* __restrict__ W2, const float* __restrict__ b2,
                const float* __restrict__ g2, const float* __restrict__ be2,
                const float* __restrict__ W3, const float* __restrict__ b3,
                float* __restrict__ out) {
    __shared__ float p[256];
    __shared__ float h1[128];
    __shared__ float h2[64];
    __shared__ float rs[2];

    const int b = blockIdx.x;
    const int t = threadIdx.x;

    p[t]       = pooled[(size_t)b * 256 + t];
    p[t + 128] = pooled[(size_t)b * 256 + t + 128];
    if (t < 2) rs[t] = 0.0f;
    __syncthreads();

    // layer 1: 256 -> 128
    float a1 = b1[t];
    {
        const float* w = W1 + (size_t)t * 256;
        for (int c = 0; c < 256; ++c) a1 += p[c] * w[c];
    }
    atomicAdd(&rs[0], a1);
    atomicAdd(&rs[1], a1 * a1);
    __syncthreads();
    {
        float mu  = rs[0] * (1.0f / 128.0f);
        float var = rs[1] * (1.0f / 128.0f) - mu * mu;
        float ninv = rsqrtf((var > 0.0f ? var : 0.0f) + 1e-5f);
        float y = (a1 - mu) * ninv * g1[t] + be1[t];
        a1 = y > 0.0f ? y : 0.01f * y;
    }
    __syncthreads();
    h1[t] = a1;
    if (t < 2) rs[t] = 0.0f;
    __syncthreads();

    // layer 2: 128 -> 64
    float a2 = 0.0f;
    if (t < 64) {
        a2 = b2[t];
        const float* w = W2 + (size_t)t * 128;
        for (int c = 0; c < 128; ++c) a2 += h1[c] * w[c];
        atomicAdd(&rs[0], a2);
        atomicAdd(&rs[1], a2 * a2);
    }
    __syncthreads();
    if (t < 64) {
        float mu  = rs[0] * (1.0f / 64.0f);
        float var = rs[1] * (1.0f / 64.0f) - mu * mu;
        float ninv = rsqrtf((var > 0.0f ? var : 0.0f) + 1e-5f);
        float y = (a2 - mu) * ninv * g2[t] + be2[t];
        h2[t] = y > 0.0f ? y : 0.01f * y;
    }
    __syncthreads();

    if (t == 0) {
        float s = b3[0];
        for (int c = 0; c < 64; ++c) s += h2[c] * W3[c];
        out[b] = s;
    }
}

// ---------------------------------------------------------------------------
// Launch
// ---------------------------------------------------------------------------
extern "C" void kernel_launch(void* const* d_in, const int* in_sizes, int n_in,
                              void* d_out, int out_size, void* d_ws, size_t ws_size,
                              hipStream_t stream) {
    (void)in_sizes; (void)n_in; (void)out_size; (void)ws_size;

    const float* trees   = (const float*)d_in[1];
    const int*   indexes = (const int*)d_in[2];
    const float* cW1 = (const float*)d_in[3];
    const float* cb1 = (const float*)d_in[4];
    const float* cW2 = (const float*)d_in[5];
    const float* cb2 = (const float*)d_in[6];
    const float* cW3 = (const float*)d_in[7];
    const float* cb3 = (const float*)d_in[8];
    const float* W1  = (const float*)d_in[9];
    const float* b1  = (const float*)d_in[10];
    const float* g1  = (const float*)d_in[11];
    const float* be1 = (const float*)d_in[12];
    const float* W2  = (const float*)d_in[13];
    const float* b2  = (const float*)d_in[14];
    const float* g2  = (const float*)d_in[15];
    const float* be2 = (const float*)d_in[16];
    const float* W3  = (const float*)d_in[17];
    const float* b3  = (const float*)d_in[18];

    unsigned char* ws = (unsigned char*)d_ws;
    bhalf* Wb1    = (bhalf*)(ws + OFF_WB1);
    bhalf* Wb2    = (bhalf*)(ws + OFF_WB2);
    bhalf* Wb3    = (bhalf*)(ws + OFF_WB3);
    float* pooled = (float*)(ws + OFF_POOL);
    bhalf* sc     = (bhalf*)(ws + OFF_SC);

    // pack conv weights into [Cout][k*Cin + c] bf16
    pack_weights<<<(512 * 128 * 3 + 255) / 256, 256, 0, stream>>>(cW1, Wb1, 512, 128);
    pack_weights<<<(512 * 512 * 3 + 255) / 256, 256, 0, stream>>>(cW2, Wb2, 512, 512);
    pack_weights<<<(256 * 512 * 3 + 255) / 256, 256, 0, stream>>>(cW3, Wb3, 256, 512);

    (void)hipFuncSetAttribute((const void*)tree_conv_fused,
                              hipFuncAttributeMaxDynamicSharedMemorySize, SMEM_BYTES);

    tree_conv_fused<<<B_SZ, NTHREADS, SMEM_BYTES, stream>>>(
        trees, indexes, cb1, cb2, cb3, Wb1, Wb2, Wb3, sc, pooled);

    mlp_kernel<<<B_SZ, 128, 0, stream>>>(pooled, W1, b1, g1, be1,
                                         W2, b2, g2, be2, W3, b3, (float*)d_out);
}
